// NonLocalMask_5076651344621
// MI455X (gfx1250) — matrix operand order repaired and checked by reading, not compiled, not run on hardware
//
#include <hip/hip_runtime.h>

// ---------- types ----------
typedef __bf16 v16bf __attribute__((ext_vector_type(16)));
typedef __bf16 v8bf  __attribute__((ext_vector_type(8)));
typedef float  v8f   __attribute__((ext_vector_type(8)));

__device__ inline __bf16 f2bf(float f) {
    union { float f; unsigned int u; } v; v.f = f;
    unsigned int r = v.u + 0x7fffu + ((v.u >> 16) & 1u);
    unsigned short h = (unsigned short)(r >> 16);
    return __builtin_bit_cast(__bf16, h);
}

__device__ inline v8f vzero8() {
    v8f z;
#pragma unroll
    for (int i = 0; i < 8; ++i) z[i] = 0.f;
    return z;
}

// A-fragment (16x32 bf16, M x K): lane holds row (lane&15),
// K = khl..khl+15 (elems 0..15), khl=(lane>>4)*16.
// `p` must already point at row*ld + kbase + khl.
__device__ inline v16bf loadA(const __bf16* p) {
    v8bf lo = *(const v8bf*)(p);
    v8bf hi = *(const v8bf*)(p + 8);
    v16bf r;
#pragma unroll
    for (int i = 0; i < 8; ++i) { r[i] = lo[i]; r[i + 8] = hi[i]; }
    return r;
}

__device__ inline float rmax16(float v) {
    v = fmaxf(v, __shfl_xor(v, 1));
    v = fmaxf(v, __shfl_xor(v, 2));
    v = fmaxf(v, __shfl_xor(v, 4));
    v = fmaxf(v, __shfl_xor(v, 8));
    return v;
}
__device__ inline float rsum16(float v) {
    v += __shfl_xor(v, 1);
    v += __shfl_xor(v, 2);
    v += __shfl_xor(v, 4);
    v += __shfl_xor(v, 8);
    return v;
}

#define WMMA_BF16(A_, B_, C_) \
    __builtin_amdgcn_wmma_f32_16x16x32_bf16(false, (A_), false, (B_), (short)0, (C_), false, false)

// ---------- conversion kernels ----------
__global__ void cvt_bf16_kernel(const float* __restrict__ in, __bf16* __restrict__ out, int n) {
    int i = blockIdx.x * 256 + threadIdx.x;
    if (i < n) out[i] = f2bf(in[i]);
}

// xt[b][n][i] = bf16(x[b][i][n]);  x viewed as (b, 256, 4096)
__global__ void transpose_cvt_kernel(const float* __restrict__ x, __bf16* __restrict__ xt) {
    __shared__ float tile[16][17];
    int b = blockIdx.z;
    int n0 = blockIdx.x * 16;
    int i0 = blockIdx.y * 16;
    const float* xp = x + (size_t)b * 1048576;
    tile[threadIdx.y][threadIdx.x] = xp[(size_t)(i0 + threadIdx.y) * 4096 + n0 + threadIdx.x];
    __syncthreads();
    __bf16* xo = xt + (size_t)b * 1048576;
    xo[(size_t)(n0 + threadIdx.y) * 256 + i0 + threadIdx.x] = f2bf(tile[threadIdx.x][threadIdx.y]);
}

// ---------- NT GEMM (bf16 in, fp32 acc, bf16 out + bias) ----------
// C[M][N] = A[M][256] * B[N][256]^T ; each wave computes a 16x64 strip.
__global__ __launch_bounds__(32) void gemm_nt_bf16_kernel(
    const __bf16* __restrict__ A, const __bf16* __restrict__ B,
    const float* __restrict__ bias, int bias_on_col,
    __bf16* __restrict__ C, int ldc) {
    int lane = threadIdx.x;
    int m0 = blockIdx.x * 16;
    int n0 = blockIdx.y * 64;
    int nlo = lane & 15;
    int khl = (lane >> 4) << 4;   // A frag K offset
    int kgB = (lane >> 4) << 4;   // B frag K offset

    v8f acc[4];
#pragma unroll
    for (int j = 0; j < 4; ++j) acc[j] = vzero8();

    const __bf16* arow = A + (size_t)(m0 + nlo) * 256 + khl;
#pragma unroll
    for (int kk = 0; kk < 8; ++kk) {
        v16bf a = loadA(arow + kk * 32);
#pragma unroll
        for (int j = 0; j < 4; ++j) {
            v16bf bfr = *(const v16bf*)(B + (size_t)(n0 + j * 16 + nlo) * 256 + kk * 32 + kgB);
            acc[j] = WMMA_BF16(a, bfr, acc[j]);
        }
    }
    int mr = (lane >> 4) << 3;
#pragma unroll
    for (int j = 0; j < 4; ++j) {
        int ncol = n0 + j * 16 + nlo;
        float bc = bias_on_col ? bias[ncol] : 0.f;
#pragma unroll
        for (int r = 0; r < 8; ++r) {
            float v = acc[j][r] + (bias_on_col ? bc : bias[m0 + mr + r]);
            C[(size_t)(m0 + mr + r) * ldc + ncol] = f2bf(v);
        }
    }
}

// ---------- flash attention ----------
// theta, phi: (b, 4096, 256) bf16 [n][o] ; gT: (b, 256, 4096) bf16 [o][n]
// Y: (b, 4096, 256) fp32 = softmax(theta phi^T) g
__global__ __launch_bounds__(32) void flash_attn_kernel(
    const __bf16* __restrict__ theta, const __bf16* __restrict__ phi,
    const __bf16* __restrict__ gT, float* __restrict__ Y) {
    __shared__ __attribute__((aligned(32))) __bf16 lds_p[16 * 32];

    int lane = threadIdx.x;
    int blk = blockIdx.x;
    int batch = blk >> 8;           // 256 query blocks per batch
    int qb = (blk & 255) << 4;
    const __bf16* th = theta + (size_t)batch * 4096 * 256;
    const __bf16* ph = phi + (size_t)batch * 4096 * 256;
    const __bf16* gp = gT + (size_t)batch * 256 * 4096;
    float* yp = Y + (size_t)batch * 4096 * 256;

    int nlo = lane & 15;
    int khl = (lane >> 4) << 4;
    int kgB = (lane >> 4) << 4;
    int mr  = (lane >> 4) << 3;

    // preload Q fragments (16 rows x 256 d)
    v16bf qf[8];
    const __bf16* qrow = th + (size_t)(qb + nlo) * 256 + khl;
#pragma unroll
    for (int kk = 0; kk < 8; ++kk) qf[kk] = loadA(qrow + kk * 32);

    v8f oacc[16];
#pragma unroll
    for (int t = 0; t < 16; ++t) oacc[t] = vzero8();
    float mrow[8], lrow[8];
#pragma unroll
    for (int r = 0; r < 8; ++r) { mrow[r] = -3.0e38f; lrow[r] = 0.f; }

    for (int k0 = 0; k0 < 4096; k0 += 32) {
        // S tiles for keys [k0, k0+16) and [k0+16, k0+32)
        v8f s0 = vzero8(), s1 = vzero8();
        const __bf16* b0 = ph + (size_t)(k0 + nlo) * 256 + kgB;
        const __bf16* b1 = ph + (size_t)(k0 + 16 + nlo) * 256 + kgB;
#pragma unroll
        for (int kk = 0; kk < 8; ++kk) {
            v16bf bf0 = *(const v16bf*)(b0 + kk * 32);
            s0 = WMMA_BF16(qf[kk], bf0, s0);
            v16bf bf1 = *(const v16bf*)(b1 + kk * 32);
            s1 = WMMA_BF16(qf[kk], bf1, s1);
        }
        // online softmax (row r lives on lanes of this 16-lane half, row id mr+r)
#pragma unroll
        for (int r = 0; r < 8; ++r) {
            float mx = rmax16(fmaxf(s0[r], s1[r]));
            float mnew = fmaxf(mrow[r], mx);
            float sc = __expf(mrow[r] - mnew);
            float p0 = __expf(s0[r] - mnew);
            float p1 = __expf(s1[r] - mnew);
            float rs = rsum16(p0 + p1);
            lrow[r] = lrow[r] * sc + rs;
            mrow[r] = mnew;
            s0[r] = p0; s1[r] = p1;
#pragma unroll
            for (int t = 0; t < 16; ++t) oacc[t][r] *= sc;
        }
        // stage P (16x32) to LDS in row-major [q][k] so it can be re-read as an A fragment
#pragma unroll
        for (int r = 0; r < 8; ++r) {
            lds_p[(mr + r) * 32 + nlo]      = f2bf(s0[r]);
            lds_p[(mr + r) * 32 + 16 + nlo] = f2bf(s1[r]);
        }
        __syncthreads();
        v16bf pa = loadA((const __bf16*)&lds_p[nlo * 32 + khl]);
        __syncthreads();
        // O += P * V  (V tile: K=keys, N=o from gT[o][n])
#pragma unroll
        for (int t = 0; t < 16; ++t) {
            v16bf vf = *(const v16bf*)(gp + (size_t)(t * 16 + nlo) * 4096 + k0 + kgB);
            oacc[t] = WMMA_BF16(pa, vf, oacc[t]);
        }
    }
    // normalize + store Y[q][o]
#pragma unroll
    for (int r = 0; r < 8; ++r) {
        float inv = 1.f / lrow[r];
#pragma unroll
        for (int t = 0; t < 16; ++t)
            yp[(size_t)(qb + mr + r) * 256 + t * 16 + nlo] = oacc[t][r] * inv;
    }
}

// ---------- mask_feat = x + gamma * (conv1x1(y_img, wws) + bws) ----------
// y_img[b][c2][p] = Y[b][p & 4095][4*c2 + (p>>12)]
__global__ void fuse_y_conv1x1_kernel(
    const float* __restrict__ Y, const float* __restrict__ x,
    const float* __restrict__ wws, const float* __restrict__ bws,
    const float* __restrict__ gamma, float* __restrict__ mask_feat) {
    int idx = blockIdx.x * 256 + threadIdx.x;
    if (idx >= 4 * 64 * 16384) return;
    int p = idx & 16383;
    int c = (idx >> 14) & 63;
    int b = idx >> 20;
    const float* yrow = Y + (size_t)b * 1048576 + (size_t)(p & 4095) * 256 + (p >> 12);
    const float* wr = wws + c * 64;
    float s = 0.f;
#pragma unroll 8
    for (int c2 = 0; c2 < 64; ++c2) s += wr[c2] * yrow[c2 * 4];
    mask_feat[idx] = x[idx] + gamma[0] * (s + bws[c]);
}

// ---------- conv3x3 64->16 + relu ----------
__global__ void conv3x3_relu_kernel(const float* __restrict__ in, const float* __restrict__ w,
                                    const float* __restrict__ bias, float* __restrict__ out) {
    int idx = blockIdx.x * 256 + threadIdx.x;
    if (idx >= 4 * 16 * 16384) return;
    int p = idx & 16383;
    int co = (idx >> 14) & 15;
    int b = idx >> 18;
    int h = p >> 7, wx = p & 127;
    const float* ip = in + (size_t)b * 64 * 16384;
    const float* wp = w + co * 64 * 9;
    float s = bias[co];
    for (int ci = 0; ci < 64; ++ci) {
        const float* ic = ip + ci * 16384;
        const float* wc = wp + ci * 9;
#pragma unroll
        for (int kh = 0; kh < 3; ++kh) {
            int hh = h + kh - 1;
            if ((unsigned)hh >= 128u) continue;
#pragma unroll
            for (int kw = 0; kw < 3; ++kw) {
                int wi = wx + kw - 1;
                if ((unsigned)wi >= 128u) continue;
                s += wc[kh * 3 + kw] * ic[hh * 128 + wi];
            }
        }
    }
    out[idx] = fmaxf(s, 0.f);
}

// ---------- conv3x3 16->1 + sigmoid ----------
__global__ void conv3x3_sigmoid_kernel(const float* __restrict__ in, const float* __restrict__ w,
                                       const float* __restrict__ bias, float* __restrict__ out) {
    int idx = blockIdx.x * 256 + threadIdx.x;
    if (idx >= 4 * 16384) return;
    int p = idx & 16383;
    int b = idx >> 14;
    int h = p >> 7, wx = p & 127;
    const float* ip = in + (size_t)b * 16 * 16384;
    float s = bias[0];
    for (int ci = 0; ci < 16; ++ci) {
        const float* ic = ip + ci * 16384;
        const float* wc = w + ci * 9;
#pragma unroll
        for (int kh = 0; kh < 3; ++kh) {
            int hh = h + kh - 1;
            if ((unsigned)hh >= 128u) continue;
#pragma unroll
            for (int kw = 0; kw < 3; ++kw) {
                int wi = wx + kw - 1;
                if ((unsigned)wi >= 128u) continue;
                s += wc[kh * 3 + kw] * ic[hh * 128 + wi];
            }
        }
    }
    out[idx] = 1.f / (1.f + __expf(-s));
}

// ---------- partial conv stage 1 (input = img * mask, mask = mb broadcast over 3ch) ----------
__global__ void pconv1_kernel(const float* __restrict__ img, const float* __restrict__ mb,
                              const float* __restrict__ w, const float* __restrict__ bias,
                              float* __restrict__ xo, float* __restrict__ mask_out) {
    const int HW = 63 * 63;
    int idx = blockIdx.x * 256 + threadIdx.x;
    if (idx >= 4 * 3 * HW) return;
    int sp = idx % HW;
    int co = (idx / HW) % 3;
    int b = idx / (3 * HW);
    int oh = sp / 63, ow = sp % 63;
    int ih = oh * 2, iw = ow * 2;
    const float* mbp = mb + (size_t)b * 16384;
    float msum = 0.f;
#pragma unroll
    for (int kh = 0; kh < 3; ++kh)
#pragma unroll
        for (int kw = 0; kw < 3; ++kw) msum += mbp[(ih + kh) * 128 + iw + kw];
    float mask_sum = 3.f * msum;
    float conv = 0.f;
    const float* ip = img + (size_t)b * 3 * 16384;
#pragma unroll
    for (int ci = 0; ci < 3; ++ci) {
        const float* ic = ip + ci * 16384;
        const float* wc = w + (co * 3 + ci) * 9;
#pragma unroll
        for (int kh = 0; kh < 3; ++kh)
#pragma unroll
            for (int kw = 0; kw < 3; ++kw) {
                float mv = mbp[(ih + kh) * 128 + iw + kw];
                conv += wc[kh * 3 + kw] * ic[(ih + kh) * 128 + iw + kw] * mv;
            }
    }
    int hole = (mask_sum == 0.f);
    xo[idx] = hole ? 0.f : (conv / mask_sum + bias[co]);
    if (co == 0) mask_out[b * HW + sp] = hole ? 0.f : 1.f;
}

// ---------- partial conv stages 2/3 (3 input channels, single-channel mask) ----------
__global__ void pconv_kernel(const float* __restrict__ xin, const float* __restrict__ min_,
                             const float* __restrict__ w, const float* __restrict__ bias,
                             float* __restrict__ xo, float* __restrict__ mout,
                             int CO, int IH, int OH) {
    int HW = OH * OH;
    int total = 4 * CO * HW;
    int idx = blockIdx.x * 256 + threadIdx.x;
    if (idx >= total) return;
    int sp = idx % HW;
    int co = (idx / HW) % CO;
    int b = idx / (CO * HW);
    int oh = sp / OH, ow = sp % OH;
    int ih = oh * 2, iw = ow * 2;
    int IHW = IH * IH;
    const float* mp = min_ + (size_t)b * IHW;
    float msum = 0.f;
#pragma unroll
    for (int kh = 0; kh < 3; ++kh)
#pragma unroll
        for (int kw = 0; kw < 3; ++kw) msum += mp[(ih + kh) * IH + iw + kw];
    float mask_sum = 3.f * msum;
    float conv = 0.f;
    const float* ip = xin + (size_t)b * 3 * IHW;
#pragma unroll
    for (int ci = 0; ci < 3; ++ci) {
        const float* ic = ip + ci * IHW;
        const float* wc = w + (co * 3 + ci) * 9;
#pragma unroll
        for (int kh = 0; kh < 3; ++kh)
#pragma unroll
            for (int kw = 0; kw < 3; ++kw)
                conv += wc[kh * 3 + kw] * ic[(ih + kh) * IH + iw + kw];
    }
    int hole = (mask_sum == 0.f);
    xo[idx] = hole ? 0.f : (conv / mask_sum + bias[co]);
    if (mout != nullptr && co == 0) mout[b * HW + sp] = hole ? 0.f : 1.f;
}

// ---------- host launch ----------
extern "C" void kernel_launch(void* const* d_in, const int* in_sizes, int n_in,
                              void* d_out, int out_size, void* d_ws, size_t ws_size,
                              hipStream_t stream) {
    const float* x    = (const float*)d_in[0];
    const float* img  = (const float*)d_in[1];
    const float* wg   = (const float*)d_in[2];
    const float* bg   = (const float*)d_in[3];
    const float* wth  = (const float*)d_in[4];
    const float* bth  = (const float*)d_in[5];
    const float* wph  = (const float*)d_in[6];
    const float* bph  = (const float*)d_in[7];
    const float* wws  = (const float*)d_in[8];
    const float* bws  = (const float*)d_in[9];
    const float* gamma= (const float*)d_in[10];
    const float* wm1  = (const float*)d_in[11];
    const float* bm1  = (const float*)d_in[12];
    const float* wm2  = (const float*)d_in[13];
    const float* bm2  = (const float*)d_in[14];
    const float* wp1  = (const float*)d_in[15];
    const float* bp1  = (const float*)d_in[16];
    const float* wp2  = (const float*)d_in[17];
    const float* bp2  = (const float*)d_in[18];
    const float* wp3  = (const float*)d_in[19];
    const float* bp3  = (const float*)d_in[20];

    float* out = (float*)d_out;
    float* out_xo = out;                        // (4,1,15,15) = 900
    float* out_mf = out + 900;                  // (4,64,128,128)
    float* out_mb = out + 900 + 4 * 64 * 16384; // (4,128,128)

    char* ws = (char*)d_ws;
    size_t off = 0;
    auto alloc = [&](size_t bytes) -> char* {
        char* p = ws + off;
        off = (off + bytes + 255) & ~(size_t)255;
        return p;
    };
    __bf16* xt    = (__bf16*)alloc((size_t)16384 * 256 * 2);
    __bf16* wgb   = (__bf16*)alloc(65536 * 2);
    __bf16* wthb  = (__bf16*)alloc(65536 * 2);
    __bf16* wphb  = (__bf16*)alloc(65536 * 2);
    __bf16* theta = (__bf16*)alloc((size_t)16384 * 256 * 2);
    __bf16* phi   = (__bf16*)alloc((size_t)16384 * 256 * 2);
    __bf16* gT    = (__bf16*)alloc((size_t)4 * 256 * 4096 * 2);
    float*  Y     = (float*)alloc((size_t)16384 * 256 * 4);
    float*  hid   = (float*)alloc((size_t)4 * 16 * 16384 * 4);
    float*  xo1   = (float*)alloc((size_t)4 * 3 * 63 * 63 * 4);
    float*  m1    = (float*)alloc((size_t)4 * 63 * 63 * 4);
    float*  xo2   = (float*)alloc((size_t)4 * 3 * 31 * 31 * 4);
    float*  m2    = (float*)alloc((size_t)4 * 31 * 31 * 4);

    // 1. weight + activation conversion
    cvt_bf16_kernel<<<256, 256, 0, stream>>>(wg, wgb, 65536);
    cvt_bf16_kernel<<<256, 256, 0, stream>>>(wth, wthb, 65536);
    cvt_bf16_kernel<<<256, 256, 0, stream>>>(wph, wphb, 65536);
    transpose_cvt_kernel<<<dim3(256, 16, 4), dim3(16, 16), 0, stream>>>(x, xt);

    // 2. projections (WMMA GEMMs)
    gemm_nt_bf16_kernel<<<dim3(1024, 4), 32, 0, stream>>>(xt, wthb, bth, 1, theta, 256);
    gemm_nt_bf16_kernel<<<dim3(1024, 4), 32, 0, stream>>>(xt, wphb, bph, 1, phi, 256);
    for (int b = 0; b < 4; ++b) {
        gemm_nt_bf16_kernel<<<dim3(16, 64), 32, 0, stream>>>(
            wgb, xt + (size_t)b * 1048576, bg, 0, gT + (size_t)b * 1048576, 4096);
    }

    // 3. flash attention (WMMA)
    flash_attn_kernel<<<1024, 32, 0, stream>>>(theta, phi, gT, Y);

    // 4. mask_feat = x + gamma * conv1x1(y_img)   -> output region
    fuse_y_conv1x1_kernel<<<(4 * 64 * 16384) / 256, 256, 0, stream>>>(Y, x, wws, bws, gamma, out_mf);

    // 5. mask head
    conv3x3_relu_kernel<<<(4 * 16 * 16384) / 256, 256, 0, stream>>>(out_mf, wm1, bm1, hid);
    conv3x3_sigmoid_kernel<<<(4 * 16384) / 256, 256, 0, stream>>>(hid, wm2, bm2, out_mb);

    // 6. partial convs
    pconv1_kernel<<<(4 * 3 * 63 * 63 + 255) / 256, 256, 0, stream>>>(img, out_mb, wp1, bp1, xo1, m1);
    pconv_kernel<<<(4 * 3 * 31 * 31 + 255) / 256, 256, 0, stream>>>(xo1, m1, wp2, bp2, xo2, m2, 3, 63, 31);
    pconv_kernel<<<(4 * 1 * 15 * 15 + 255) / 256, 256, 0, stream>>>(xo2, m2, wp3, bp3, out_xo, nullptr, 1, 31, 15);

    (void)in_sizes; (void)n_in; (void)out_size; (void)ws_size;
}
